// MultiHeadAttention_31267361915364
// MI455X (gfx1250) — matrix operand order repaired and checked by reading, not compiled, not run on hardware
//
#include <hip/hip_runtime.h>

// ---------------- types / helpers ----------------
typedef __attribute__((ext_vector_type(16))) __bf16    bf16x16;
typedef __attribute__((ext_vector_type(8)))  float     floatx8;
typedef __attribute__((ext_vector_type(4)))  unsigned  u32x4;

union Frag {
    bf16x16        v;
    u32x4          q[2];
    unsigned       u[8];
    unsigned short h[16];
};

static __device__ inline unsigned short f2bf(float f) {
    unsigned u = __float_as_uint(f);
    unsigned r = u + 0x7FFFu + ((u >> 16) & 1u);   // round-to-nearest-even
    return (unsigned short)(r >> 16);
}

static __device__ inline floatx8 wmma_bf16(const Frag& a, const Frag& b, floatx8 c) {
    return __builtin_amdgcn_wmma_f32_16x16x32_bf16(false, a.v, false, b.v,
                                                   (short)0, c, false, false);
}

#define SEQ_T  2048
#define DIM_C  1024
#define HEADS  16
#define DHEAD  64

// ---------------- f32 -> bf16 conversion (2-wide) ----------------
__global__ void cvt_bf16x2(const float* __restrict__ src, unsigned* __restrict__ dst, int n2) {
    int i = blockIdx.x * blockDim.x + threadIdx.x;
    int stride = gridDim.x * blockDim.x;
    for (; i < n2; i += stride) {
        float2 f = ((const float2*)src)[i];
        dst[i] = (unsigned)f2bf(f.x) | ((unsigned)f2bf(f.y) << 16);
    }
}

// ---------------- weight reorder: f32 [K=1024][N] -> bf16 fragment-major ----------------
// Fragment (nt, kt) = 16(n) x 32(k) tile; lane l: n = nt*16 + (l&15),
// K base = (l<16 ? 0 : 8); dword i packs K = kt*32 + base + (i<4 ? 2i : 16 + 2(i-4)) (+1).
// Wp dword index o = ((nt*32 + kt)*32 + lane)*8 + i   (KT = 1024/32 = 32)
__global__ void reorder_w(const float* __restrict__ W, unsigned* __restrict__ Wp,
                          int N, int total) {
    int o = blockIdx.x * blockDim.x + threadIdx.x;
    int stride = gridDim.x * blockDim.x;
    for (; o < total; o += stride) {
        int i    = o & 7;
        int lane = (o >> 3) & 31;
        int kt   = (o >> 8) & 31;
        int nt   = o >> 13;
        int n    = nt * 16 + (lane & 15);
        int k    = kt * 32 + ((lane < 16) ? 0 : 8) + ((i < 4) ? 2 * i : 16 + 2 * (i - 4));
        unsigned lo16 = f2bf(W[(size_t)k * N + n]);
        unsigned hi16 = f2bf(W[(size_t)(k + 1) * N + n]);
        Wp[o] = lo16 | (hi16 << 16);
    }
}

// ---------------- GEMM core: LDS-free, direct b128 fragment loads ----------------
// Block: 256 threads = 8 waves; wave computes 32(M) x 64(N); block = 256 x 64.
#define BM 256
#define BN 64
#define KT_STRIDE 32   // K tiles per column of fragments (K=1024 / 32)

// QKV: [8192,1024] x [1024,3072] + bias -> K/Q/V [B,H,T,d] bf16 (split order k,q,v)
__global__ __launch_bounds__(256) void gemm_qkv(
        const unsigned short* __restrict__ A,    // x bf16 [8192][1024] row-major
        const unsigned*       __restrict__ Bp,   // W_kqv fragment-major
        const float*          __restrict__ bias, // [3072]
        unsigned short* __restrict__ Kt,
        unsigned short* __restrict__ Qt,
        unsigned short* __restrict__ Vt) {
    const int wave = threadIdx.x >> 5;
    const int lane = threadIdx.x & 31;
    const int lo   = lane & 15;
    const int akb  = (lane < 16) ? 0 : 8;
    const int rowBase = blockIdx.x * BM + wave * 32;
    const int colBase = blockIdx.y * BN;
    const int ntBase  = colBase >> 4;

    floatx8 acc[2][4] = {};

    for (int kt = 0; kt < KT_STRIDE; kt++) {
        Frag a[2];
        for (int mt = 0; mt < 2; mt++) {
            const unsigned short* ar = A + (size_t)(rowBase + mt * 16 + lo) * DIM_C + kt * 32 + akb;
            a[mt].q[0] = *(const u32x4*)(ar);
            a[mt].q[1] = *(const u32x4*)(ar + 16);
        }
        for (int nt = 0; nt < 4; nt++) {
            const unsigned* bp = Bp + ((size_t)((ntBase + nt) * KT_STRIDE + kt) * 32 + lane) * 8;
            Frag b;
            b.q[0] = *(const u32x4*)(bp);
            b.q[1] = *(const u32x4*)(bp + 4);
            acc[0][nt] = wmma_bf16(a[0], b, acc[0][nt]);
            acc[1][nt] = wmma_bf16(a[1], b, acc[1][nt]);
        }
    }

    const int chunk = colBase >> 10;
    unsigned short* dst = (chunk == 0) ? Kt : ((chunk == 1) ? Qt : Vt);
    const int cB = colBase & 1023;
    for (int nt = 0; nt < 4; nt++) {
        int col = cB + nt * 16 + lo;
        float bv = bias[colBase + nt * 16 + lo];
        int h = col >> 6, dd = col & 63;
        for (int mt = 0; mt < 2; mt++)
            for (int r = 0; r < 8; r++) {
                int R = rowBase + mt * 16 + r + ((lane < 16) ? 0 : 8);
                int b = R >> 11, t = R & 2047;
                dst[((size_t)(b * HEADS + h) * SEQ_T + t) * DHEAD + dd] =
                    f2bf(acc[mt][nt][r] + bv);
            }
    }
}

// Output projection: [8192,1024] x [1024,1024] + bias -> f32 [8192][1024]
__global__ __launch_bounds__(256) void gemm_out(
        const unsigned short* __restrict__ A,    // attn out bf16 [8192][1024]
        const unsigned*       __restrict__ Bp,   // W_out fragment-major
        const float*          __restrict__ bias, // [1024]
        float* __restrict__ out) {
    const int wave = threadIdx.x >> 5;
    const int lane = threadIdx.x & 31;
    const int lo   = lane & 15;
    const int akb  = (lane < 16) ? 0 : 8;
    const int rowBase = blockIdx.x * BM + wave * 32;
    const int colBase = blockIdx.y * BN;
    const int ntBase  = colBase >> 4;

    floatx8 acc[2][4] = {};

    for (int kt = 0; kt < KT_STRIDE; kt++) {
        Frag a[2];
        for (int mt = 0; mt < 2; mt++) {
            const unsigned short* ar = A + (size_t)(rowBase + mt * 16 + lo) * DIM_C + kt * 32 + akb;
            a[mt].q[0] = *(const u32x4*)(ar);
            a[mt].q[1] = *(const u32x4*)(ar + 16);
        }
        for (int nt = 0; nt < 4; nt++) {
            const unsigned* bp = Bp + ((size_t)((ntBase + nt) * KT_STRIDE + kt) * 32 + lane) * 8;
            Frag b;
            b.q[0] = *(const u32x4*)(bp);
            b.q[1] = *(const u32x4*)(bp + 4);
            acc[0][nt] = wmma_bf16(a[0], b, acc[0][nt]);
            acc[1][nt] = wmma_bf16(a[1], b, acc[1][nt]);
        }
    }

    for (int nt = 0; nt < 4; nt++) {
        int col = colBase + nt * 16 + lo;
        float bv = bias[col];
        for (int mt = 0; mt < 2; mt++)
            for (int r = 0; r < 8; r++) {
                int R = rowBase + mt * 16 + r + ((lane < 16) ? 0 : 8);
                out[(size_t)R * DIM_C + col] = acc[mt][nt][r] + bv;
            }
    }
}

// ---------------- Flash attention: causal, scale = 1/(d/2) = 1/32 ----------------
__global__ __launch_bounds__(256) void attn_flash(
        const unsigned short* __restrict__ Qt,
        const unsigned short* __restrict__ Kt,
        const unsigned short* __restrict__ Vt,
        unsigned short* __restrict__ Ot) {   // [B,T,C] bf16 (heads concatenated)
    __shared__ __align__(16) unsigned short ldsP[8][16][32];  // per-wave P staging, 8 KB
    const int wave = threadIdx.x >> 5;
    const int lane = threadIdx.x & 31;
    const int lo   = lane & 15;
    const bool hi  = lane >= 16;
    const int hi8  = hi ? 8 : 0;

    const int gw = blockIdx.x * 8 + wave;  // global 16-row q-block id [0, 8192)
    const int qb = gw & 127;
    const int bh = gw >> 7;                // [0, 64)
    const int b  = bh >> 4;
    const int h  = bh & 15;

    const unsigned short* Qg = Qt + ((size_t)bh * SEQ_T + qb * 16) * DHEAD;
    const unsigned short* Kg = Kt + (size_t)bh * SEQ_T * DHEAD;
    const unsigned short* Vg = Vt + (size_t)bh * SEQ_T * DHEAD;

    // Q A-fragments (K-chunks 0..31, 32..63): contiguous b128 loads
    Frag aQ0, aQ1;
    const int akb = hi ? 16 : 0;
    {
        const unsigned short* qr = Qg + lo * DHEAD + akb;
        aQ0.q[0] = *(const u32x4*)(qr);
        aQ0.q[1] = *(const u32x4*)(qr + 8);
        aQ1.q[0] = *(const u32x4*)(qr + 32);
        aQ1.q[1] = *(const u32x4*)(qr + 40);
    }

    floatx8 O0 = {}, O1 = {}, O2 = {}, O3 = {};
    const float NEG_INF = -__builtin_inff();
    float mrow[8], lrow[8];
    for (int r = 0; r < 8; r++) { mrow[r] = NEG_INF; lrow[r] = 0.0f; }

    const float scale = 1.0f / 32.0f;           // 1/(d/2), faithful to reference
    const int nkb = (qb * 16 + 15) / 32 + 1;    // causal key-block count (32 keys each)
    const int bkb = hi ? 16 : 0;                // B-fragment K-group base

    for (int jb = 0; jb < nkb; jb++) {
        if (jb + 1 < nkb) {
            __builtin_prefetch(Kg + (size_t)(jb + 1) * 32 * DHEAD, 0, 0);
            __builtin_prefetch(Vg + (size_t)(jb + 1) * 32 * DHEAD, 0, 0);
        }

        // S = Q * K^T  (two 16x16 N-tiles, K-dim 64 = 2 WMMA each)
        floatx8 S0 = {}, S1 = {};
        {
            const unsigned short* k0r = Kg + (size_t)(jb * 32 + lo) * DHEAD + bkb;
            const unsigned short* k1r = k0r + (size_t)16 * DHEAD;
            Frag bk;
            bk.q[0] = *(const u32x4*)(k0r);      bk.q[1] = *(const u32x4*)(k0r + 8);
            S0 = wmma_bf16(aQ0, bk, S0);
            bk.q[0] = *(const u32x4*)(k0r + 32); bk.q[1] = *(const u32x4*)(k0r + 40);
            S0 = wmma_bf16(aQ1, bk, S0);
            bk.q[0] = *(const u32x4*)(k1r);      bk.q[1] = *(const u32x4*)(k1r + 8);
            S1 = wmma_bf16(aQ0, bk, S1);
            bk.q[0] = *(const u32x4*)(k1r + 32); bk.q[1] = *(const u32x4*)(k1r + 40);
            S1 = wmma_bf16(aQ1, bk, S1);
        }

        // online softmax (row r lives on vgpr r, lanes 0-15 / 16-31 split)
        const int key0 = jb * 32 + lo;
        const int key1 = jb * 32 + 16 + lo;
        float p0v[8], p1v[8], alpha[8];
        for (int r = 0; r < 8; r++) {
            int rowg = qb * 16 + r + hi8;
            float s0 = S0[r] * scale; if (key0 > rowg) s0 = NEG_INF;
            float s1 = S1[r] * scale; if (key1 > rowg) s1 = NEG_INF;
            float mx = fmaxf(s0, s1);
            for (int mm = 1; mm < 16; mm <<= 1) mx = fmaxf(mx, __shfl_xor(mx, mm, 32));
            float mnew = fmaxf(mrow[r], mx);
            float al = __expf(mrow[r] - mnew);
            float p0 = __expf(s0 - mnew);
            float p1 = __expf(s1 - mnew);
            float rs = p0 + p1;
            for (int mm = 1; mm < 16; mm <<= 1) rs += __shfl_xor(rs, mm, 32);
            lrow[r] = lrow[r] * al + rs;
            mrow[r] = mnew;
            alpha[r] = al; p0v[r] = p0; p1v[r] = p1;
        }
        for (int r = 0; r < 8; r++) {
            O0[r] *= alpha[r]; O1[r] *= alpha[r];
            O2[r] *= alpha[r]; O3[r] *= alpha[r];
        }

        // re-layout P: C-format -> LDS -> A-format (in-wave; LDS is in-order per wave)
        for (int r = 0; r < 8; r++) {
            ldsP[wave][r + hi8][lo]      = f2bf(p0v[r]);
            ldsP[wave][r + hi8][16 + lo] = f2bf(p1v[r]);
        }
        asm volatile("s_wait_dscnt 0" ::: "memory");
        Frag aP;
        aP.q[0] = *(const u32x4*)&ldsP[wave][lo][akb];
        aP.q[1] = *(const u32x4*)&ldsP[wave][lo][akb + 8];

        // O += P * V   (4 dim-chunks of 16, K = 32 keys)
        for (int c = 0; c < 4; c++) {
            Frag bv;
            for (int i = 0; i < 8; i++) {
                int k = bkb + 2 * i;
                unsigned h0 = Vg[(size_t)(jb * 32 + k)     * DHEAD + c * 16 + lo];
                unsigned h1 = Vg[(size_t)(jb * 32 + k + 1) * DHEAD + c * 16 + lo];
                bv.u[i] = h0 | (h1 << 16);
            }
            floatx8* Oc = (c == 0) ? &O0 : (c == 1) ? &O1 : (c == 2) ? &O2 : &O3;
            *Oc = wmma_bf16(aP, bv, *Oc);
        }
    }

    // normalize + store: Ot[b, t, h*64 + dim]  (bf16)
    for (int r = 0; r < 8; r++) {
        int rowg = qb * 16 + r + hi8;
        float inv = 1.0f / lrow[r];
        size_t base = ((size_t)b * SEQ_T + rowg) * DIM_C + h * DHEAD;
        Ot[base + 0 * 16 + lo] = f2bf(O0[r] * inv);
        Ot[base + 1 * 16 + lo] = f2bf(O1[r] * inv);
        Ot[base + 2 * 16 + lo] = f2bf(O2[r] * inv);
        Ot[base + 3 * 16 + lo] = f2bf(O3[r] * inv);
    }
}

// ---------------- launcher ----------------
extern "C" void kernel_launch(void* const* d_in, const int* in_sizes, int n_in,
                              void* d_out, int out_size, void* d_ws, size_t ws_size,
                              hipStream_t stream) {
    (void)in_sizes; (void)n_in; (void)out_size; (void)ws_size;
    const float* x      = (const float*)d_in[0];   // [4,2048,1024]
    const float* W_kqv  = (const float*)d_in[1];   // [1024,3072]
    const float* b_kqv  = (const float*)d_in[2];   // [3072]
    const float* W_out  = (const float*)d_in[3];   // [1024,1024]
    const float* b_out  = (const float*)d_in[4];   // [1024]
    float* out = (float*)d_out;

    char* ws = (char*)d_ws;
    const size_t MB = (size_t)1 << 20;
    unsigned short* xb     = (unsigned short*)(ws);             // 16 MB
    unsigned*       WkqvP  = (unsigned*)      (ws + 16 * MB);   //  6 MB (fragment-major)
    unsigned*       WoutP  = (unsigned*)      (ws + 22 * MB);   //  2 MB (fragment-major)
    unsigned short* Kt     = (unsigned short*)(ws + 24 * MB);   // 16 MB
    unsigned short* Qt     = (unsigned short*)(ws + 40 * MB);   // 16 MB
    unsigned short* Vt     = (unsigned short*)(ws + 56 * MB);   // 16 MB
    unsigned short* Ot     = (unsigned short*)(ws + 72 * MB);   // 16 MB (total 88 MB)

    // 1. conversions / weight swizzle
    cvt_bf16x2<<<4096, 256, 0, stream>>>(x, (unsigned*)xb, 8192 * 1024 / 2);
    reorder_w <<<3072, 256, 0, stream>>>(W_kqv, WkqvP, 3072, 1024 * 3072 / 2);
    reorder_w <<<1024, 256, 0, stream>>>(W_out, WoutP, 1024, 1024 * 1024 / 2);

    // 2. QKV projection (fused bias + [B,H,T,d] scatter)
    gemm_qkv<<<dim3(8192 / BM, 3072 / BN), 256, 0, stream>>>(xb, WkqvP, b_kqv, Kt, Qt, Vt);

    // 3. causal flash attention (8192 q-waves, 8 per block)
    attn_flash<<<1024, 256, 0, stream>>>(Qt, Kt, Vt, Ot);

    // 4. output projection (f32 out + bias)
    gemm_out<<<dim3(8192 / BM, 1024 / BN), 256, 0, stream>>>(Ot, WoutP, b_out, out);
}
